// Transformer_49323404427975
// MI455X (gfx1250) — matrix operand order repaired and checked by reading, not compiled, not run on hardware
//
#include <hip/hip_runtime.h>
#include <hip/hip_bf16.h>

// ---------------- problem constants (from reference) ----------------
#define NN    20000
#define EE    320000
#define DIN   256
#define HID   128
#define NHEAD 4
#define DOUT  40

typedef __attribute__((ext_vector_type(16))) _Float16 v16h;
typedef __attribute__((ext_vector_type(8)))  float    v8f;

// ---------------- helpers ----------------
__device__ __forceinline__ unsigned f2ord(float f) {
  unsigned u = __float_as_uint(f);
  return (u & 0x80000000u) ? ~u : (u | 0x80000000u);
}
__device__ __forceinline__ float ord2f(unsigned u) {
  return (u & 0x80000000u) ? __uint_as_float(u & 0x7fffffffu)
                           : __uint_as_float(~u);
}

__global__ void cvt_f32_f16_kernel(const float* __restrict__ in,
                                   _Float16* __restrict__ out, size_t n) {
  size_t i = (size_t)blockIdx.x * blockDim.x + threadIdx.x;
  size_t stride = (size_t)gridDim.x * blockDim.x;
  for (; i < n; i += stride) out[i] = (_Float16)in[i];
}

__global__ void fill_u32_kernel(unsigned* __restrict__ p, unsigned val, size_t n) {
  size_t i = (size_t)blockIdx.x * blockDim.x + threadIdx.x;
  size_t stride = (size_t)gridDim.x * blockDim.x;
  for (; i < n; i += stride) p[i] = val;
}

// ---------------- B-fragment packing ----------------
// Packs f32 weight W[K,N] (row-major) into per-wave WMMA B fragments (f16):
//   P[tile][kstep][lane][0..15]  (16 halves = 32B contiguous per lane)
// B-fragment order used here (contiguous k, same as A): lane&15 = column within tile;
// lanes 0-15 hold K k0+0..15, lanes 16-31 hold K k0+16..31.
// Tiles beyond N are zero-filled, so the GEMM needs no load guards.
__global__ void pack_b_kernel(const float* __restrict__ W,
                              _Float16* __restrict__ P,
                              int K, int N, int ntp) {
  int idx = blockIdx.x * blockDim.x + threadIdx.x;   // one thread = one lane-fragment
  int total = ntp * (K >> 5) * 32;
  if (idx >= total) return;
  int lane  = idx & 31;
  int t     = idx >> 5;
  int ks    = K >> 5;
  int kstep = t % ks;
  int tile  = t / ks;
  int half  = lane >> 4;
  int l     = lane & 15;
  int col   = tile * 16 + l;
  int kbase = kstep * 32 + half * 16;
  v16h out;
#pragma unroll
  for (int j = 0; j < 16; ++j)
    out[j] = (col < N) ? (_Float16)W[(size_t)(kbase + j) * N + col] : (_Float16)0.0f;
  *(v16h*)(P + (size_t)idx * 16) = out;
}

// ---------------- WMMA GEMM: C[M,N] = A[M,K](f16) @ W + bias ----------------
// One wave computes a 16 x 32 output block (2 WMMA accumulators, shared A frag).
// A fragment order used here (contiguous k): lanes 0-15 row M=lane hold
// K {0..15}; lanes 16-31 row M=lane-16 hold K {16..31} (matches packed B).
// Requires: M % 16 == 0, K % 32 == 0, ntp (padded B tiles) % 2 == 0.
__global__ void wmma_gemm_packed_kernel(const _Float16* __restrict__ A,
                                        const _Float16* __restrict__ Bp,
                                        const float* __restrict__ bias,
                                        float* __restrict__ C,
                                        int M, int K, int N, int ntp) {
  const int NSUB = 2;
  int wave = (blockIdx.x * blockDim.x + threadIdx.x) >> 5;
  int ngrp = ntp / NSUB;
  int tm   = (wave / ngrp) << 4;
  int grp  = wave % ngrp;
  if (tm >= M) return;                  // uniform per wave -> EXEC stays all-ones
  int lane = threadIdx.x & 31;
  int half = lane >> 4;
  int l    = lane & 15;
  int ks   = K >> 5;

  v8f acc0 = {}, acc1 = {};
  const _Float16* Ap  = A + (size_t)(tm + l) * K + half * 16;
  const _Float16* Bp0 = Bp + (((size_t)(grp * NSUB + 0) * ks) * 32 + lane) * 16;
  const _Float16* Bp1 = Bp + (((size_t)(grp * NSUB + 1) * ks) * 32 + lane) * 16;

  for (int kstep = 0; kstep < ks; ++kstep) {
    v16h a;
    const _Float16* ap = Ap + kstep * 32;
    __builtin_prefetch(ap + 32, 0, 0);               // global_prefetch_b8
#pragma unroll
    for (int i = 0; i < 8; ++i) {
      a[i]     = ap[i];
      a[8 + i] = ap[8 + i];
    }
    v16h b0 = *(const v16h*)(Bp0 + (size_t)kstep * 512);  // 32 lanes * 16 halves
    v16h b1 = *(const v16h*)(Bp1 + (size_t)kstep * 512);
    acc0 = __builtin_amdgcn_wmma_f32_16x16x32_f16(false, a, false, b0,
                                                  (short)0, acc0, false, false);
    acc1 = __builtin_amdgcn_wmma_f32_16x16x32_f16(false, a, false, b1,
                                                  (short)0, acc1, false, false);
  }

  // D layout (f32 16x16): lane&15 = col; VGPR r -> row r (lanes 0-15) / 8+r (16-31)
  int row0 = tm + half * 8;
#pragma unroll
  for (int s = 0; s < NSUB; ++s) {
    int col = (grp * NSUB + s) * 16 + l;
    if (col < N) {
      float bv = bias[col];
      const v8f& acc = s ? acc1 : acc0;
#pragma unroll
      for (int r = 0; r < 8; ++r)
        C[(size_t)(row0 + r) * N + col] = acc[r] + bv;
    }
  }
}

// ---------------- edge attention kernels ----------------
__global__ void edge_logits_kernel(const float* __restrict__ q,
                                   const float* __restrict__ k,
                                   const long long* __restrict__ src,
                                   const long long* __restrict__ dst,
                                   float* __restrict__ alpha,
                                   unsigned* __restrict__ segmax,
                                   int C, float scale) {
  int idx = blockIdx.x * blockDim.x + threadIdx.x;
  if (idx >= EE * NHEAD) return;
  int e = idx >> 2, h = idx & 3;
  int sN = (int)src[e], dN = (int)dst[e];
  const float4* qp = (const float4*)(q + ((size_t)dN * NHEAD + h) * C);
  const float4* kp = (const float4*)(k + ((size_t)sN * NHEAD + h) * C);
  float acc = 0.f;
  int c4 = C >> 2;
  for (int c = 0; c < c4; ++c) {
    float4 a4 = qp[c], b4 = kp[c];
    acc += a4.x * b4.x + a4.y * b4.y + a4.z * b4.z + a4.w * b4.w;
  }
  acc *= scale;
  alpha[idx] = acc;
  atomicMax(&segmax[(size_t)dN * NHEAD + h], f2ord(acc));
}

__global__ void edge_expsum_kernel(float* __restrict__ alpha,
                                   const unsigned* __restrict__ segmax,
                                   float* __restrict__ ssum,
                                   const long long* __restrict__ dst) {
  int idx = blockIdx.x * blockDim.x + threadIdx.x;
  if (idx >= EE * NHEAD) return;
  int e = idx >> 2, h = idx & 3;
  int dN = (int)dst[e];
  float m = ord2f(segmax[(size_t)dN * NHEAD + h]);
  float ee = __expf(alpha[idx] - m);
  alpha[idx] = ee;
  atomicAdd(&ssum[(size_t)dN * NHEAD + h], ee);
}

__global__ void edge_agg_kernel(const float* __restrict__ alpha,
                                const float* __restrict__ ssum,
                                const float* __restrict__ v,
                                const long long* __restrict__ src,
                                const long long* __restrict__ dst,
                                float* __restrict__ agg, int C) {
  int idx = blockIdx.x * blockDim.x + threadIdx.x;
  if (idx >= EE * NHEAD) return;
  int e = idx >> 2, h = idx & 3;
  int sN = (int)src[e], dN = (int)dst[e];
  float a = alpha[idx] / (ssum[(size_t)dN * NHEAD + h] + 1e-16f);
  const float* vp = v + ((size_t)sN * NHEAD + h) * C;
  float* ap = agg + ((size_t)dN * NHEAD + h) * C;
  for (int c = 0; c < C; c += 4) {
    atomicAdd(&ap[c + 0], vp[c + 0] * a);
    atomicAdd(&ap[c + 1], vp[c + 1] * a);
    atomicAdd(&ap[c + 2], vp[c + 2] * a);
    atomicAdd(&ap[c + 3], vp[c + 3] * a);
  }
}

__global__ void mean_skip_relu_kernel(const float* __restrict__ agg,
                                      const float* __restrict__ skip,
                                      float* __restrict__ out, int C) {
  int idx = blockIdx.x * blockDim.x + threadIdx.x;
  if (idx >= NN * C) return;
  int n = idx / C, c = idx - n * C;
  float s = 0.f;
#pragma unroll
  for (int h = 0; h < NHEAD; ++h) s += agg[((size_t)n * NHEAD + h) * C + c];
  float y = s * (1.0f / NHEAD) + skip[(size_t)n * C + c];
  out[idx] = fmaxf(y, 0.f);
}

__global__ void final_kernel(const float* __restrict__ agg,
                             const float* __restrict__ skip,
                             float* __restrict__ out) {
  int n = blockIdx.x * blockDim.x + threadIdx.x;
  if (n >= NN) return;
  float y[DOUT];
  float mx = -INFINITY;
#pragma unroll 4
  for (int c = 0; c < DOUT; ++c) {
    float s = 0.f;
#pragma unroll
    for (int h = 0; h < NHEAD; ++h) s += agg[((size_t)n * NHEAD + h) * DOUT + c];
    float v = fmaxf(s * (1.0f / NHEAD) + skip[(size_t)n * DOUT + c], 0.f);
    y[c] = v;
    mx = fmaxf(mx, v);
  }
  float se = 0.f;
#pragma unroll 4
  for (int c = 0; c < DOUT; ++c) se += __expf(y[c] - mx);
  float lse = mx + __logf(se);
#pragma unroll 4
  for (int c = 0; c < DOUT; ++c) out[(size_t)n * DOUT + c] = y[c] - lse;
}

// ---------------- host-side helpers ----------------
static inline int cdiv(int a, int b) { return (a + b - 1) / b; }

static void launch_gemm(const _Float16* A, const _Float16* Bp, const float* bias,
                        float* C, int M, int K, int N, int ntp, hipStream_t s) {
  int waves = (M / 16) * (ntp / 2);
  int blocks = cdiv(waves, 8);                    // 8 waves (256 threads) per block
  wmma_gemm_packed_kernel<<<blocks, 256, 0, s>>>(A, Bp, bias, C, M, K, N, ntp);
}

static void launch_pack(const float* W, _Float16* P, int K, int N, int ntp,
                        hipStream_t s) {
  int total = ntp * (K / 32) * 32;
  pack_b_kernel<<<cdiv(total, 256), 256, 0, s>>>(W, P, K, N, ntp);
}

static void launch_cvt(const float* in, _Float16* out, size_t n, hipStream_t s) {
  int blocks = (int)((n + 255) / 256); if (blocks > 4096) blocks = 4096;
  cvt_f32_f16_kernel<<<blocks, 256, 0, s>>>(in, out, n);
}

static void launch_fill(void* p, unsigned val, size_t n, hipStream_t s) {
  int blocks = (int)((n + 255) / 256); if (blocks > 4096) blocks = 4096;
  fill_u32_kernel<<<blocks, 256, 0, s>>>((unsigned*)p, val, n);
}

static void run_attention(const float* qb, const float* kb, const float* vb,
                          const long long* src, const long long* dst,
                          float* alphab, unsigned* mb, float* sb, float* aggb,
                          int C, hipStream_t s) {
  size_t nh = (size_t)NN * NHEAD;
  launch_fill(mb, 0x007FFFFFu, nh, s);            // ordered-uint encoding of -inf
  launch_fill(sb, 0u, nh, s);
  launch_fill(aggb, 0u, (size_t)NN * NHEAD * C, s);
  int eh = EE * NHEAD;
  float scale = 1.0f / sqrtf((float)C);
  edge_logits_kernel<<<cdiv(eh, 256), 256, 0, s>>>(qb, kb, src, dst, alphab, mb, C, scale);
  edge_expsum_kernel<<<cdiv(eh, 256), 256, 0, s>>>(alphab, mb, sb, dst);
  edge_agg_kernel<<<cdiv(eh, 256), 256, 0, s>>>(alphab, sb, vb, src, dst, aggb, C);
}

extern "C" void kernel_launch(void* const* d_in, const int* in_sizes, int n_in,
                              void* d_out, int out_size, void* d_ws, size_t ws_size,
                              hipStream_t stream) {
  (void)in_sizes; (void)n_in; (void)out_size; (void)ws_size;
  const float*     x   = (const float*)d_in[0];
  const long long* ei  = (const long long*)d_in[1];   // int64 [2,E]
  const long long* src = ei;
  const long long* dst = ei + EE;

  // p0: d_in[2..9], p1: d_in[10..17], p2: d_in[18..25]
  const float* W[3][4]; const float* Bz[3][4];
  for (int L = 0; L < 3; ++L)
    for (int j = 0; j < 4; ++j) {
      W[L][j]  = (const float*)d_in[2 + L * 8 + 2 * j];
      Bz[L][j] = (const float*)d_in[2 + L * 8 + 2 * j + 1];
    }

  // -------- workspace carve-up --------
  char* ws = (char*)d_ws; size_t off = 0;
  auto alloc = [&](size_t bytes) -> void* {
    void* p = ws + off; off += (bytes + 255) & ~(size_t)255; return p;
  };
  float*    qb    = (float*)alloc((size_t)NN * NHEAD * HID * 4);
  float*    kb    = (float*)alloc((size_t)NN * NHEAD * HID * 4);
  float*    vb    = (float*)alloc((size_t)NN * NHEAD * HID * 4);
  float*    aggb  = (float*)alloc((size_t)NN * NHEAD * HID * 4);
  float*    skipb = (float*)alloc((size_t)NN * HID * 4);
  float*    hbuf  = (float*)alloc((size_t)NN * HID * 4);
  float*    alphab= (float*)alloc((size_t)EE * NHEAD * 4);
  unsigned* mb    = (unsigned*)alloc((size_t)NN * NHEAD * 4);
  float*    sb    = (float*)alloc((size_t)NN * NHEAD * 4);
  _Float16* xh    = (_Float16*)alloc((size_t)NN * DIN * 2);   // reused as f16 activation
  _Float16* wh    = (_Float16*)alloc((size_t)1000000 * 2);    // packed f16 B fragments

  // per-weight geometry: K, N, padded tile count (multiple of NSUB=2)
  const int Kdim[3] = { DIN, HID, HID };
  const int Cdim[3] = { HID, HID, DOUT };
  int  Ndim[3][4], Ntp[3][4];
  size_t woff[3][4]; size_t w = 0;
  for (int L = 0; L < 3; ++L) {
    int HC = NHEAD * Cdim[L];
    for (int j = 0; j < 4; ++j) {
      int N = (j < 3) ? HC : Cdim[L];
      int ntp = cdiv(N, 16); ntp += (ntp & 1);    // pad to even
      Ndim[L][j] = N; Ntp[L][j] = ntp;
      woff[L][j] = w;
      w += (size_t)ntp * (Kdim[L] / 32) * 512;    // 32 lanes * 16 halves per kstep
    }
  }

  // -------- convert activation, pack weights into WMMA B fragments --------
  launch_cvt(x, xh, (size_t)NN * DIN, stream);
  for (int L = 0; L < 3; ++L)
    for (int j = 0; j < 4; ++j)
      launch_pack(W[L][j], wh + woff[L][j], Kdim[L], Ndim[L][j], Ntp[L][j], stream);

  float* dstbuf[4];
  for (int L = 0; L < 3; ++L) {
    int K = Kdim[L], C = Cdim[L];
    dstbuf[0] = qb; dstbuf[1] = kb; dstbuf[2] = vb; dstbuf[3] = skipb;
    for (int j = 0; j < 4; ++j)
      launch_gemm(xh, wh + woff[L][j], Bz[L][j], dstbuf[j],
                  NN, K, Ndim[L][j], Ntp[L][j], stream);
    // sparse attention (gathers hit L2: q/k/v ~41MB each << 192MB)
    run_attention(qb, kb, vb, src, dst, alphab, mb, sb, aggb, C, stream);
    if (L < 2) {
      mean_skip_relu_kernel<<<cdiv(NN * C, 256), 256, 0, stream>>>(aggb, skipb, hbuf, C);
      launch_cvt(hbuf, xh, (size_t)NN * C, stream);   // next layer's f16 input
    } else {
      final_kernel<<<cdiv(NN, 256), 256, 0, stream>>>(aggb, skipb, (float*)d_out);
    }
  }
}
